// SlidingWindowProcessor_39745627357262
// MI455X (gfx1250) — matrix operand order repaired and checked by reading, not compiled, not run on hardware
//
#include <hip/hip_runtime.h>
#include <math.h>

typedef __attribute__((ext_vector_type(16))) _Float16 v16h;
typedef __attribute__((ext_vector_type(8)))  float    v8f;

#define B_  8
#define T_  2048
#define C_  128
#define H_  256
#define W_  16
#define G3  (3*H_)          // 768
#define TP  (T_ + W_ - 1)   // 2063

// ---- packed-weight fragment layout in workspace -------------------------
// One WMMA B-fragment (f16 32x16) = 32 lanes x 16 halves, stored so that
// lane L's 16 halves are contiguous: element e of lane L = W[N][K] with
//   K = kc*32 + (e<8 ? (L>>4)*8 + e : 16 + (L>>4)*8 + (e-8)),  N = nt*16 + (L&15)
#define FRAG_HALVES 512
#define NFRAG_IH    (48*4)   // N=768 -> 48 tiles, K=128 -> 4 chunks
#define NFRAG_HH    (48*8)   // N=768, K=256
#define NFRAG_PROJ  (16*8)   // N=256, K=256
#define OFF_IH      0
#define OFF_HH      (NFRAG_IH*FRAG_HALVES)
#define OFF_PROJ    (OFF_HH + NFRAG_HH*FRAG_HALVES)
#define HALVES_TOTAL (OFF_PROJ + NFRAG_PROJ*FRAG_HALVES)

#define XI_ROWS_RAW (B_*TP)                    // 16504
#define XI_TILES    ((XI_ROWS_RAW + 15)/16)    // 1032
#define XI_BYTE_OFF (((size_t)HALVES_TOTAL*2 + 255) & ~(size_t)255)

#define MTILE     32    // rows of (b,t) per block in the GRU kernel
#define H_STRIDE  264   // halves: 528B row, not a multiple of 256B (banks)
#define Y_STRIDE  260   // floats: 1040B row, not a multiple of 256B

static __device__ __forceinline__ v8f wmma_f16(v16h a, v16h b, v8f c) {
  return __builtin_amdgcn_wmma_f32_16x16x32_f16(false, a, false, b,
                                                (short)0, c, false, false);
}

// f16 16x32 A-fragment per ISA layout: lane (kh = lane>>4, row = lane&15),
// VGPR v<4 holds K pairs 2v,2v+1 (+kh*8); v>=4 holds 16+2(v-4) (+kh*8).
static __device__ __forceinline__ v16h load_a(const _Float16* hrow, int kh, int kc) {
  v16h a;
#pragma unroll
  for (int v = 0; v < 8; ++v) {
    int pb = (v < 4 ? 2 * v : 16 + 2 * (v - 4)) + kh * 8;
    int k  = kc * 32 + pb;
    a[2 * v]     = hrow[k];
    a[2 * v + 1] = hrow[k + 1];
  }
  return a;
}

// ---------------- kernel 0: pack weights to f16 B-fragments ---------------
__global__ void pack_weights_kernel(const float* __restrict__ w_ih,
                                    const float* __restrict__ w_hh,
                                    const float* __restrict__ w_proj,
                                    _Float16* __restrict__ wpack) {
  int f    = blockIdx.x;
  int lane = threadIdx.x;        // 32 threads = 1 wave
  int nl = lane & 15, kh = lane >> 4;
  const float* src;
  int nt, kc, kdim, base;
  if (f < NFRAG_IH)                  { src = w_ih;   nt = f/4; kc = f%4; kdim = C_; base = OFF_IH   + f*FRAG_HALVES; }
  else if (f < NFRAG_IH + NFRAG_HH)  { int g = f - NFRAG_IH;            src = w_hh;   nt = g/8; kc = g%8; kdim = H_; base = OFF_HH   + g*FRAG_HALVES; }
  else                               { int g = f - NFRAG_IH - NFRAG_HH; src = w_proj; nt = g/8; kc = g%8; kdim = H_; base = OFF_PROJ + g*FRAG_HALVES; }
  int N = nt * 16 + nl;
#pragma unroll
  for (int e = 0; e < 16; ++e) {
    int K = kc * 32 + (e < 8 ? kh * 8 + e : 16 + kh * 8 + (e - 8));
    wpack[base + lane * 16 + e] = (_Float16)src[(size_t)N * kdim + K];
  }
}

// ---------------- kernel 1: xi = x_pad @ w_ih^T + b_ih  (fp32 out) --------
__global__ __launch_bounds__(512)
void xi_kernel(const float* __restrict__ x, const float* __restrict__ b_ih,
               const _Float16* __restrict__ wpack, float* __restrict__ xi) {
  int tid  = threadIdx.x;
  int lane = tid & 31, wave = tid >> 5;
  int nl = lane & 15, kh = lane >> 4;
  int mt = blockIdx.x;

  int m_global = mt * 16 + nl;                 // padded row id (A rows)
  bool valid = m_global < XI_ROWS_RAW;
  int b  = valid ? m_global / TP : 0;
  int tt = valid ? m_global % TP : 0;
  bool have = valid && (tt >= (W_ - 1));
  int   xr    = have ? (b * T_ + tt - (W_ - 1)) : 0;   // always-safe row
  float scale = have ? 1.0f : 0.0f;                    // zero out pad rows
  const float* xrow = x + (size_t)xr * C_;

  int nt0 = wave * 3;                          // 16 waves x 3 = 48 N-tiles
  v8f c0 = {}, c1 = {}, c2 = {};
#pragma unroll
  for (int kc = 0; kc < 4; ++kc) {             // K = 128
    v16h a;
#pragma unroll
    for (int v = 0; v < 8; ++v) {
      int pb = (v < 4 ? 2 * v : 16 + 2 * (v - 4)) + kh * 8;
      int k  = kc * 32 + pb;
      a[2 * v]     = (_Float16)(xrow[k]     * scale);
      a[2 * v + 1] = (_Float16)(xrow[k + 1] * scale);
    }
    const v16h* bp0 = (const v16h*)(wpack + OFF_IH + (size_t)((nt0 + 0) * 4 + kc) * FRAG_HALVES + lane * 16);
    const v16h* bp1 = (const v16h*)(wpack + OFF_IH + (size_t)((nt0 + 1) * 4 + kc) * FRAG_HALVES + lane * 16);
    const v16h* bp2 = (const v16h*)(wpack + OFF_IH + (size_t)((nt0 + 2) * 4 + kc) * FRAG_HALVES + lane * 16);
    c0 = wmma_f16(a, *bp0, c0);
    c1 = wmma_f16(a, *bp1, c1);
    c2 = wmma_f16(a, *bp2, c2);
  }
  int rb = kh * 8;
#pragma unroll
  for (int i = 0; i < 8; ++i) {
    size_t row = (size_t)(mt * 16 + rb + i) * G3;
    int g0 = (nt0 + 0) * 16 + nl;
    int g1 = (nt0 + 1) * 16 + nl;
    int g2 = (nt0 + 2) * 16 + nl;
    xi[row + g0] = c0[i] + b_ih[g0];
    xi[row + g1] = c1[i] + b_ih[g1];
    xi[row + g2] = c2[i] + b_ih[g2];
  }
}

// -------- kernel 2: fused 16-step GRU scan + proj + LayerNorm + GELU ------
// Wave j owns gate N-tiles {j, 16+j, 32+j}: r/z/n accumulators for the SAME
// 16 hidden columns with identical C-lane layouts -> gate math stays in
// registers, h carried in fp32 registers, only the f16 A-copy goes via LDS.
__global__ __launch_bounds__(512)
void gru_kernel(const float* __restrict__ xi, const float* __restrict__ b_hh,
                const float* __restrict__ b_proj, const float* __restrict__ ln_g,
                const float* __restrict__ ln_b, const _Float16* __restrict__ wpack,
                float* __restrict__ out) {
  __shared__ _Float16 h_lds[MTILE * H_STRIDE];   // f16 h (WMMA A source)
  __shared__ float    y_lds[MTILE * Y_STRIDE];   // projection result for LN

  int tid  = threadIdx.x;
  int lane = tid & 31, wave = tid >> 5;          // 16 waves
  int nl = lane & 15, kh = lane >> 4;

  int m0 = blockIdx.x * MTILE;                   // MTILE consecutive (b,t)
  int b  = m0 / T_;                              // T_ % MTILE == 0
  int t0 = m0 % T_;
  size_t xi_row0 = (size_t)b * TP + t0;

  for (int i = tid; i < MTILE * H_STRIDE; i += 512) h_lds[i] = (_Float16)0.0f;
  __syncthreads();

  const _Float16* hrow0 = &h_lds[nl * H_STRIDE];          // M-subtile 0
  const _Float16* hrow1 = &h_lds[(16 + nl) * H_STRIDE];   // M-subtile 1

  int cg = wave * 16 + nl;                       // this lane's hidden column
  float bhr = b_hh[cg];
  float bhz = b_hh[H_ + cg];
  float bhn = b_hh[2 * H_ + cg];

  float hreg[2][8];                              // fp32 h at (subtile, i)
#pragma unroll
  for (int s = 0; s < 2; ++s)
#pragma unroll
    for (int i = 0; i < 8; ++i) hreg[s][i] = 0.0f;

  for (int w = 0; w < W_; ++w) {
    // ---- gh = h @ w_hh^T : [32x256]*[256x768] ----
    v8f cr[2] = {{}, {}}, cz[2] = {{}, {}}, cn[2] = {{}, {}};
#pragma unroll
    for (int kc = 0; kc < 8; ++kc) {
      v16h a0 = load_a(hrow0, kh, kc);
      v16h a1 = load_a(hrow1, kh, kc);
      const v16h* bpr = (const v16h*)(wpack + OFF_HH + (size_t)((wave      ) * 8 + kc) * FRAG_HALVES + lane * 16);
      const v16h* bpz = (const v16h*)(wpack + OFF_HH + (size_t)((16 + wave) * 8 + kc) * FRAG_HALVES + lane * 16);
      const v16h* bpn = (const v16h*)(wpack + OFF_HH + (size_t)((32 + wave) * 8 + kc) * FRAG_HALVES + lane * 16);
      v16h br = *bpr, bz = *bpz, bn = *bpn;
      cr[0] = wmma_f16(a0, br, cr[0]);  cr[1] = wmma_f16(a1, br, cr[1]);
      cz[0] = wmma_f16(a0, bz, cz[0]);  cz[1] = wmma_f16(a1, bz, cz[1]);
      cn[0] = wmma_f16(a0, bn, cn[0]);  cn[1] = wmma_f16(a1, bn, cn[1]);
    }
    __syncthreads();                             // all A-reads done

    // ---- gate update, fully in registers ----
#pragma unroll
    for (int s = 0; s < 2; ++s) {
#pragma unroll
      for (int i = 0; i < 8; ++i) {
        int row = s * 16 + kh * 8 + i;           // local (b,t) row 0..31
        size_t xir = (xi_row0 + row + w) * (size_t)G3;
        float gir = xi[xir + cg];
        float giz = xi[xir + H_ + cg];
        float gin = xi[xir + 2 * H_ + cg];
        float r = 1.0f / (1.0f + __expf(-(gir + cr[s][i] + bhr)));
        float z = 1.0f / (1.0f + __expf(-(giz + cz[s][i] + bhz)));
        float n = tanhf(gin + r * (cn[s][i] + bhn));
        float hn = (1.0f - z) * n + z * hreg[s][i];
        hreg[s][i] = hn;
        h_lds[row * H_STRIDE + cg] = (_Float16)hn;
        if (w + 1 < W_)                          // hide next step's gi latency
          __builtin_prefetch(&xi[(xi_row0 + row + w + 1) * (size_t)G3 + cg], 0, 1);
      }
    }
    __syncthreads();                             // h visible to all A-reads
  }

  // ---- projection: y = h @ w_proj^T + b_proj ; wave j -> N-tile j ----
  {
    v8f c0 = {}, c1 = {};
#pragma unroll
    for (int kc = 0; kc < 8; ++kc) {
      v16h a0 = load_a(hrow0, kh, kc);
      v16h a1 = load_a(hrow1, kh, kc);
      const v16h* bp = (const v16h*)(wpack + OFF_PROJ + (size_t)(wave * 8 + kc) * FRAG_HALVES + lane * 16);
      v16h bw = *bp;
      c0 = wmma_f16(a0, bw, c0);
      c1 = wmma_f16(a1, bw, c1);
    }
    float bp = b_proj[cg];
#pragma unroll
    for (int i = 0; i < 8; ++i) {
      y_lds[(kh * 8 + i) * Y_STRIDE + cg]        = c0[i] + bp;
      y_lds[(16 + kh * 8 + i) * Y_STRIDE + cg]   = c1[i] + bp;
    }
  }
  __syncthreads();

  // ---- LayerNorm + exact GELU: wave j owns rows j and 16+j ----
#pragma unroll
  for (int rr = 0; rr < 2; ++rr) {
    int m = wave + rr * 16;
    float v[8], s = 0.0f, s2 = 0.0f;
#pragma unroll
    for (int k = 0; k < 8; ++k) {
      float y = y_lds[m * Y_STRIDE + lane * 8 + k];
      v[k] = y; s += y; s2 += y * y;
    }
#pragma unroll
    for (int off = 16; off > 0; off >>= 1) {
      s  += __shfl_xor(s,  off, 32);
      s2 += __shfl_xor(s2, off, 32);
    }
    float mu  = s * (1.0f / H_);
    float var = s2 * (1.0f / H_) - mu * mu;
    float rs  = rsqrtf(var + 1e-5f);
    size_t ob = (size_t)(m0 + m) * H_;
#pragma unroll
    for (int k = 0; k < 8; ++k) {
      int cidx = lane * 8 + k;
      float y = (v[k] - mu) * rs * ln_g[cidx] + ln_b[cidx];
      out[ob + cidx] = 0.5f * y * (1.0f + erff(y * 0.70710678118654752f));
    }
  }
}

extern "C" void kernel_launch(void* const* d_in, const int* in_sizes, int n_in,
                              void* d_out, int out_size, void* d_ws, size_t ws_size,
                              hipStream_t stream) {
  const float* x      = (const float*)d_in[0];
  const float* w_ih   = (const float*)d_in[1];
  const float* w_hh   = (const float*)d_in[2];
  const float* b_ih   = (const float*)d_in[3];
  const float* b_hh   = (const float*)d_in[4];
  const float* w_proj = (const float*)d_in[5];
  const float* b_proj = (const float*)d_in[6];
  const float* ln_g   = (const float*)d_in[7];
  const float* ln_b   = (const float*)d_in[8];

  _Float16* wpack = (_Float16*)d_ws;
  float*    xi    = (float*)((char*)d_ws + XI_BYTE_OFF);

  pack_weights_kernel<<<NFRAG_IH + NFRAG_HH + NFRAG_PROJ, 32, 0, stream>>>(w_ih, w_hh, w_proj, wpack);
  xi_kernel<<<XI_TILES, 512, 0, stream>>>(x, b_ih, wpack, xi);
  gru_kernel<<<(B_ * T_) / MTILE, 512, 0, stream>>>(xi, b_hh, b_proj, ln_g, ln_b, wpack, (float*)d_out);
}
